// Attention_15590731285136
// MI455X (gfx1250) — matrix operand order repaired and checked by reading, not compiled, not run on hardware
//
#include <hip/hip_runtime.h>
#include <hip/hip_bf16.h>

typedef __attribute__((ext_vector_type(16))) _Float16 v16h;
typedef __attribute__((ext_vector_type(8)))  float    v8f;
typedef __attribute__((ext_vector_type(4)))  int      v4i;

#define B_   64
#define H_   2048
#define D_   512
#define HID_ 512
#define MASK_FILL_F (-4294967295.0f)

// ---------------------------------------------------------------------------
// Async global -> LDS copy (CDNA5, tracked by ASYNCcnt)
// Builtin signature (probed from diagnostic): (v4i AS1*, v4i AS3*, imm, imm)
// ---------------------------------------------------------------------------
__device__ __forceinline__ void async_ld_b128(const void* g, void* l) {
#if __has_builtin(__builtin_amdgcn_global_load_async_to_lds_b128)
  __builtin_amdgcn_global_load_async_to_lds_b128(
      (__attribute__((address_space(1))) v4i*)g,
      (__attribute__((address_space(3))) v4i*)l, 0, 0);
#else
  unsigned loff = (unsigned)(unsigned long)(__attribute__((address_space(3))) void*)l;
  unsigned long long ga = (unsigned long long)g;
  asm volatile("global_load_async_to_lds_b128 %0, %1, off"
               :: "v"(loff), "v"(ga) : "memory");
#endif
}

__device__ __forceinline__ void wait_async0() {
#if __has_builtin(__builtin_amdgcn_s_wait_asynccnt)
  __builtin_amdgcn_s_wait_asynccnt(0);
#else
  asm volatile("s_wait_asynccnt 0x0" ::: "memory");
#endif
}

// ---------------------------------------------------------------------------
// gate[b,d] = K[b,d] * W[d,d]   (einsum 'bhd,dd,bd' only uses diag(W))
// ---------------------------------------------------------------------------
__global__ void k_gate(const float* __restrict__ K, const float* __restrict__ W,
                       float* __restrict__ gate) {
  int i = blockIdx.x * 256 + threadIdx.x;
  if (i < B_ * D_) {
    int d = i & (D_ - 1);
    gate[i] = K[i] * W[d * (D_ + 1)];
  }
}

// ---------------------------------------------------------------------------
// Repack w1 (f32 [d,hid] row-major) into f16 WMMA B-fragment layout:
//   w1p[((nt*16 + kt)*32 + lane)*16 + i]
// B 32x16 f16 layout: lane n (0-15) = column n, K=0..7,16..23 ; lane n+16 = K=8..15,24..31
// ---------------------------------------------------------------------------
__global__ void k_packw1(const float* __restrict__ w1, _Float16* __restrict__ w1p) {
  int q = blockIdx.x * 256 + threadIdx.x;
  if (q < D_ * HID_) {
    int i    =  q        & 15;
    int lane = (q >> 4)  & 31;
    int kt   = (q >> 9)  & 15;
    int nt   =  q >> 13;
    int n = nt * 16 + (lane & 15);
    int k = kt * 32 + (lane >> 4) * 8 + (i & 7) + (i >> 3) * 16;
    w1p[q] = (_Float16)w1[k * HID_ + n];
  }
}

// ---------------------------------------------------------------------------
// score[b,h] = relu((V[b,h,:]*gate[b,:]) @ w1 + b1) @ w2 + b2, then mask fill.
// One wave per 16-row h-strip; A fragments (gated V, f16) stay resident in
// VGPRs. w1 fragments double-buffered in LDS, staged with async global->LDS
// loads overlapped with the WMMA phase of the previous chunk.
// ---------------------------------------------------------------------------
#define CVT4(f, off, A, G)                                                  \
  f[(off) + 0] = (_Float16)(A.x * G.x); f[(off) + 1] = (_Float16)(A.y * G.y); \
  f[(off) + 2] = (_Float16)(A.z * G.z); f[(off) + 3] = (_Float16)(A.w * G.w);

__device__ __forceinline__ void stage_chunk(const _Float16* __restrict__ w1p,
                                            _Float16* lbuf, int c, int tid) {
  const _Float16* src = w1p + (size_t)c * 16384;
#pragma unroll
  for (int t = 0; t < 8; ++t) {
    const int e = (tid + t * 256) * 8;   // 16B granules, 32KB per chunk
    async_ld_b128(src + e, lbuf + e);
  }
}

__global__ __launch_bounds__(256)
void k_score(const float* __restrict__ V, const float* __restrict__ gate,
             const _Float16* __restrict__ w1p, const float* __restrict__ b1,
             const float* __restrict__ w2, const float* __restrict__ b2,
             const int* __restrict__ mask, float* __restrict__ score) {
  __shared__ __align__(32) _Float16 lds_b[2][16384];  // 2 x 32 KB double buffer

  const int lane = threadIdx.x & 31;
  const int wv   = threadIdx.x >> 5;
  const int b    = blockIdx.y;
  const int h0   = blockIdx.x * 128 + wv * 16;
  const int lr   = lane & 15;
  const int sel  = lane >> 4;

  // Kick off chunk 0 staging; its latency is hidden by A-fragment build.
  stage_chunk(w1p, lds_b[0], 0, threadIdx.x);

  // --- Build A fragments (16-bit A 16x32 striping):
  // lane<16: row=lane, elems = K k0..k0+7 and k0+16..k0+23 with k0=kt*32
  // lane>=16: same rows, k0=kt*32+8
  const float* Vrow = V + ((size_t)b * H_ + (size_t)(h0 + lr)) * D_;
  const float* grow = gate + b * D_;
  v16h afrag[16];
#pragma unroll
  for (int kt = 0; kt < 16; ++kt) {
    const int k0 = kt * 32 + sel * 8;
    float4 a0 = *(const float4*)(Vrow + k0);
    float4 a1 = *(const float4*)(Vrow + k0 + 4);
    float4 a2 = *(const float4*)(Vrow + k0 + 16);
    float4 a3 = *(const float4*)(Vrow + k0 + 20);
    float4 g0 = *(const float4*)(grow + k0);
    float4 g1 = *(const float4*)(grow + k0 + 4);
    float4 g2 = *(const float4*)(grow + k0 + 16);
    float4 g3 = *(const float4*)(grow + k0 + 20);
    v16h f;
    CVT4(f, 0,  a0, g0)
    CVT4(f, 4,  a1, g1)
    CVT4(f, 8,  a2, g2)
    CVT4(f, 12, a3, g3)
    afrag[kt] = f;
  }

  float scorep[8] = {0.f, 0.f, 0.f, 0.f, 0.f, 0.f, 0.f, 0.f};
  const float b2v = b2[0];

  wait_async0();
  __syncthreads();

  // 16 chunks x 2 N-tiles = 32 N-tiles (hid=512)
  for (int c = 0; c < 16; ++c) {
    const int buf = c & 1;
    if (c + 1 < 16)  // prefetch next chunk into the other buffer (async)
      stage_chunk(w1p, lds_b[buf ^ 1], c + 1, threadIdx.x);

    v8f acc0, acc1;
    const float bv0 = b1[c * 32 + lr];
    const float bv1 = b1[c * 32 + 16 + lr];
#pragma unroll
    for (int i = 0; i < 8; ++i) { acc0[i] = bv0; acc1[i] = bv1; }

    const v16h* lb = (const v16h*)lds_b[buf];
#pragma unroll
    for (int kt = 0; kt < 16; ++kt) {
      v16h bf0 = lb[kt * 32 + lane];
      v16h bf1 = lb[(16 + kt) * 32 + lane];
      acc0 = __builtin_amdgcn_wmma_f32_16x16x32_f16(false, afrag[kt], false, bf0,
                                                    (short)0, acc0, false, false);
      acc1 = __builtin_amdgcn_wmma_f32_16x16x32_f16(false, afrag[kt], false, bf1,
                                                    (short)0, acc1, false, false);
    }

    // Fused epilogue: relu then dot with w2 column (N = c*32 + t*16 + lr)
    const float w20 = w2[c * 32 + lr];
    const float w21 = w2[c * 32 + 16 + lr];
#pragma unroll
    for (int i = 0; i < 8; ++i)
      scorep[i] += fmaxf(acc0[i], 0.f) * w20 + fmaxf(acc1[i], 0.f) * w21;

    wait_async0();   // next chunk's async loads landed in LDS
    __syncthreads(); // everyone done reading buf / writing buf^1
  }

  // Reduce over the 16 lanes that share each C-tile row (M = i + sel*8)
#pragma unroll
  for (int i = 0; i < 8; ++i) {
    float s = scorep[i];
    s += __shfl_xor(s, 1, 32);
    s += __shfl_xor(s, 2, 32);
    s += __shfl_xor(s, 4, 32);
    s += __shfl_xor(s, 8, 32);
    scorep[i] = s;
  }
  if (lr == 0) {  // lanes 0 and 16 write rows 0..7 and 8..15
#pragma unroll
    for (int i = 0; i < 8; ++i) {
      int h = h0 + sel * 8 + i;
      float s = scorep[i] + b2v;
      if (mask[b * H_ + h] != 0) s = MASK_FILL_F;
      score[b * H_ + h] = s;
    }
  }
}

// ---------------------------------------------------------------------------
// alpha[b,:] = softmax(score[b,:]) over h (2048 per b)
// ---------------------------------------------------------------------------
__global__ __launch_bounds__(256)
void k_softmax(const float* __restrict__ score, float* __restrict__ alpha) {
  __shared__ float red[256];
  const int b = blockIdx.x, t = threadIdx.x;
  float v[8];
  float m = -3.0e38f;
#pragma unroll
  for (int i = 0; i < 8; ++i) {
    v[i] = score[b * H_ + t + i * 256];
    m = fmaxf(m, v[i]);
  }
  red[t] = m; __syncthreads();
  for (int s = 128; s > 0; s >>= 1) {
    if (t < s) red[t] = fmaxf(red[t], red[t + s]);
    __syncthreads();
  }
  m = red[0]; __syncthreads();
  float sum = 0.f;
#pragma unroll
  for (int i = 0; i < 8; ++i) { v[i] = __expf(v[i] - m); sum += v[i]; }
  red[t] = sum; __syncthreads();
  for (int s = 128; s > 0; s >>= 1) {
    if (t < s) red[t] += red[t + s];
    __syncthreads();
  }
  const float inv = 1.0f / red[0];
#pragma unroll
  for (int i = 0; i < 8; ++i) alpha[b * H_ + t + i * 256] = v[i] * inv;
}

// ---------------------------------------------------------------------------
// out[b,d] = sum_h alpha[b,h] * V[b,h,d]   (memory-bound, coalesced)
// ---------------------------------------------------------------------------
__global__ __launch_bounds__(512)
void k_out(const float* __restrict__ V, const float* __restrict__ alpha,
           float* __restrict__ out) {
  __shared__ float as_[H_];
  const int b = blockIdx.x, t = threadIdx.x;
  for (int i = t; i < H_; i += 512) as_[i] = alpha[b * H_ + i];
  __syncthreads();
  const float* Vb = V + (size_t)b * H_ * D_ + t;
  float acc0 = 0.f, acc1 = 0.f, acc2 = 0.f, acc3 = 0.f;
  for (int h = 0; h < H_; h += 4) {
    acc0 += as_[h + 0] * Vb[(size_t)(h + 0) * D_];
    acc1 += as_[h + 1] * Vb[(size_t)(h + 1) * D_];
    acc2 += as_[h + 2] * Vb[(size_t)(h + 2) * D_];
    acc3 += as_[h + 3] * Vb[(size_t)(h + 3) * D_];
  }
  out[b * D_ + t] = (acc0 + acc1) + (acc2 + acc3);
}

// ---------------------------------------------------------------------------
extern "C" void kernel_launch(void* const* d_in, const int* in_sizes, int n_in,
                              void* d_out, int out_size, void* d_ws, size_t ws_size,
                              hipStream_t stream) {
  const float* K    = (const float*)d_in[0];
  const float* V    = (const float*)d_in[1];
  const int*   mask = (const int*)d_in[2];
  const float* W    = (const float*)d_in[3];
  const float* w1   = (const float*)d_in[4];
  const float* b1   = (const float*)d_in[5];
  const float* w2   = (const float*)d_in[6];
  const float* b2   = (const float*)d_in[7];
  float* out = (float*)d_out;

  char* ws = (char*)d_ws;
  float*    gate  = (float*)ws;                    // 128 KB
  _Float16* w1p   = (_Float16*)(ws + (128 << 10)); // 512 KB
  float*    score = (float*)(ws + (640 << 10));    // 512 KB
  float*    alpha = (float*)(ws + (1152 << 10));   // 512 KB

  k_gate   <<<dim3((B_ * D_ + 255) / 256), dim3(256), 0, stream>>>(K, W, gate);
  k_packw1 <<<dim3((D_ * HID_ + 255) / 256), dim3(256), 0, stream>>>(w1, w1p);
  k_score  <<<dim3(H_ / 128, B_), dim3(256), 0, stream>>>(V, gate, w1p, b1, w2, b2, mask, score);
  k_softmax<<<dim3(B_), dim3(256), 0, stream>>>(score, alpha);
  k_out    <<<dim3(B_), dim3(512), 0, stream>>>(V, alpha, out);
}
